// ACNN_35072702939144
// MI455X (gfx1250) — compile-verified, matrix operand order checked by reading, not run on hardware
//
#include <hip/hip_runtime.h>
#include <hip/hip_bf16.h>

// ---------------------------------------------------------------------------
// MI455X (gfx1250) implementation.
//  * FC head (18.4 GFLOP of the ~35 GFLOP total) runs on v_wmma_f32_16x16x32_f16.
//  * W tiles for the GEMM are staged in LDS with global_load_async_to_lds_b128
//    (ASYNCcnt-tracked) so the 8 waves of a block share one copy and the copy
//    overlaps the WMMAs of the current chunk.
//  * Convs are direct, LDS-staged per sample (C_in too small for useful WMMA K).
//  * Everything fits in the 192MB L2; scratch usage ~161MB.
// ---------------------------------------------------------------------------

typedef _Float16 v16h __attribute__((ext_vector_type(16)));
typedef _Float16 v8h  __attribute__((ext_vector_type(8)));
typedef float    v8f  __attribute__((ext_vector_type(8)));

#define BATCH 2048

// ---------------------------------------------------------------- converts
__global__ void f32_to_f16_kernel(const float* __restrict__ src,
                                  _Float16* __restrict__ dst, int n) {
    int i = blockIdx.x * blockDim.x + threadIdx.x;
    if (i < n) dst[i] = (_Float16)src[i];
}

// Wf4 is [10,64]; pad to [16,64] with zero rows so the WMMA N-tile is full.
__global__ void pack_w4_kernel(const float* __restrict__ w,
                               _Float16* __restrict__ dst) {
    int i = blockIdx.x * blockDim.x + threadIdx.x;
    if (i < 16 * 64) {
        int n = i >> 6;
        dst[i] = (n < 10) ? (_Float16)w[i] : (_Float16)0.0f;
    }
}

// ------------------------------------------------- conv 1->16, k3, s1, relu
__global__ void conv3x3_s1_kernel(const float* __restrict__ X,
                                  const float* __restrict__ w,
                                  const float* __restrict__ bias,
                                  float* __restrict__ out) {
    __shared__ float xs[784];
    __shared__ float wl[144];
    __shared__ float bl[16];
    const int b = blockIdx.x, tid = threadIdx.x;
    for (int i = tid; i < 784; i += blockDim.x) xs[i] = X[(size_t)b * 784 + i];
    if (tid < 144) wl[tid] = w[tid];
    if (tid < 16)  bl[tid] = bias[tid];
    __syncthreads();
    for (int p = tid; p < 676; p += blockDim.x) {
        const int oy = p / 26, ox = p - oy * 26;
        float acc[16];
#pragma unroll
        for (int o = 0; o < 16; ++o) acc[o] = bl[o];
#pragma unroll
        for (int ky = 0; ky < 3; ++ky)
#pragma unroll
            for (int kx = 0; kx < 3; ++kx) {
                const float xv = xs[(oy + ky) * 28 + ox + kx];
#pragma unroll
                for (int o = 0; o < 16; ++o)
                    acc[o] = fmaf(xv, wl[o * 9 + ky * 3 + kx], acc[o]);
            }
#pragma unroll
        for (int o = 0; o < 16; ++o)
            out[((size_t)b * 16 + o) * 676 + p] = fmaxf(acc[o], 0.0f);
    }
}

// ------------------------------------- generic conv k5 s2 + relu (direct)
// one block per sample; input tile + weights staged in dynamic LDS
// (94.5KB or 220KB -- within the 320KB/WGP LDS of CDNA5).
__global__ void conv5x5_s2_kernel(const float* __restrict__ in,
                                  const float* __restrict__ w,
                                  const float* __restrict__ bias,
                                  float* __restrict__ out,
                                  int cin, int hin, int cout, int hout) {
    extern __shared__ float smem[];
    const int b = blockIdx.x, tid = threadIdx.x;
    const int in_elems = cin * hin * hin;
    const int w_elems  = cout * cin * 25;
    float* xsm = smem;
    float* wsm = smem + in_elems;
    const float* inb = in + (size_t)b * in_elems;
    for (int i = tid; i < in_elems; i += blockDim.x) xsm[i] = inb[i];
    for (int i = tid; i < w_elems;  i += blockDim.x) wsm[i] = w[i];
    __syncthreads();

    const int P = hout * hout;
    const int groups = cout >> 3;            // 8 output channels per work item
    const int work = groups * P;
    for (int it = tid; it < work; it += blockDim.x) {
        const int g = it / P, p = it - g * P;
        const int oy = p / hout, ox = p - oy * hout;
        const int oc0 = g * 8;
        float acc[8];
#pragma unroll
        for (int o = 0; o < 8; ++o) acc[o] = bias[oc0 + o];
        for (int ic = 0; ic < cin; ++ic) {
            const float* xrow = xsm + ic * hin * hin + (oy * 2) * hin + ox * 2;
            const float* wrow = wsm + (oc0 * cin + ic) * 25;
            const int wstride = cin * 25;
#pragma unroll
            for (int ky = 0; ky < 5; ++ky)
#pragma unroll
                for (int kx = 0; kx < 5; ++kx) {
                    const float xv = xrow[ky * hin + kx];
                    const int wi = ky * 5 + kx;
#pragma unroll
                    for (int o = 0; o < 8; ++o)
                        acc[o] = fmaf(xv, wrow[o * wstride + wi], acc[o]);
                }
        }
#pragma unroll
        for (int o = 0; o < 8; ++o)
            out[((size_t)b * cout + oc0 + o) * P + p] = fmaxf(acc[o], 0.0f);
    }
}

// ---------------------------------------------- channel sum of out1 [B,32,121]
__global__ void chansum_kernel(const float* __restrict__ out1,
                               float* __restrict__ s) {
    int idx = blockIdx.x * blockDim.x + threadIdx.x;
    if (idx >= BATCH * 121) return;
    const int b = idx / 121, p = idx - b * 121;
    const float* base = out1 + (size_t)b * 32 * 121 + p;
    float a = 0.0f;
#pragma unroll
    for (int c = 0; c < 32; ++c) a += base[c * 121];
    s[idx] = a;
}

// --------------- out3[b,j,y,x] = sum_k s[b,y+dy,x+dx]*out2[b,j,dy,dx] -> f16
__global__ void crosscorr_kernel(const float* __restrict__ s,
                                 const float* __restrict__ out2,
                                 _Float16* __restrict__ f0) {
    __shared__ float sl[121];
    __shared__ float o2[1024];
    const int b = blockIdx.x, tid = threadIdx.x;
    for (int i = tid; i < 121;  i += blockDim.x) sl[i] = s[(size_t)b * 121 + i];
    for (int i = tid; i < 1024; i += blockDim.x) o2[i] = out2[(size_t)b * 1024 + i];
    __syncthreads();
    for (int idx = tid; idx < 4096; idx += blockDim.x) {
        const int j = idx >> 6, yx = idx & 63, y = yx >> 3, x = yx & 7;
        float a = 0.0f;
#pragma unroll
        for (int dy = 0; dy < 4; ++dy)
#pragma unroll
            for (int dx = 0; dx < 4; ++dx)
                a = fmaf(sl[(y + dy) * 11 + (x + dx)], o2[j * 16 + dy * 4 + dx], a);
        f0[(size_t)b * 4096 + idx] = (_Float16)a;
    }
}

// ------------------------------------------------------------ WMMA GEMM
// C[M,N] = A[M,K] * W[N,K]^T + bias, optional ReLU+f16 store (else f32).
// blockDim = 256 (8 waves); wave w owns rows m0 = bx*128 + w*16, block owns
// cols n0 = by*16*NT .. +16*NT (all waves share the same W columns).
// Per 32-wide K-chunk:
//   * the [16*NT x 32] W chunk is copied global->LDS with one
//     global_load_async_to_lds_b128 per thread (ASYNCcnt-tracked),
//   * each wave pulls its 4 B fragments with ds_load_b128 pairs,
//   * the next chunk's async copy is issued after the reads (single 4KB
//     buffer, two barriers), overlapping the copy with the 4 WMMAs.
// Fragment layout per CDNA5 ISA: lane = (half<<4)|row, element e of v16h is
// K = (e/8)*16 + half*8 + e%8 -> two contiguous 16B reads per fragment.
template <int NT, bool RELU_F16>
__global__ void gemm_wmma_kernel(const _Float16* __restrict__ A,
                                 const _Float16* __restrict__ W,
                                 const float* __restrict__ bias, int bias_n,
                                 void* __restrict__ out, int N, int K) {
    __shared__ __align__(16) _Float16 wtile[NT * 16 * 32];
    const int tid  = threadIdx.x;
    const int wave = tid >> 5;
    const int lane = tid & 31;
    const int half = lane >> 4;
    const int mr   = lane & 15;
    const int m0 = blockIdx.x * 128 + wave * 16;
    const int n0 = blockIdx.y * (16 * NT);

    // async producer mapping: thread t -> W row r = t>>2, 16B slice = t&3
    const int r    = tid >> 2;
    const int part = tid & 3;
    const bool producer = (tid < NT * 64);
    const _Float16* wsrc = W + (size_t)(n0 + (producer ? r : 0)) * K + part * 8;
    const unsigned  wdst =
        (unsigned)(uintptr_t)(&wtile[0]) + (unsigned)(r * 64 + part * 16);

    v8f zero = {};
    v8f acc[NT];
#pragma unroll
    for (int j = 0; j < NT; ++j) acc[j] = zero;

    const _Float16* arow = A + (size_t)(m0 + mr) * K + half * 8;

    if (producer)
        asm volatile("global_load_async_to_lds_b128 %0, %1, off"
                     :: "v"(wdst), "v"(wsrc) : "memory");

    for (int k0 = 0; k0 < K; k0 += 32) {
        // A fragment straight from global (unique rows per wave, L2-resident)
        const v8h alo = *(const v8h*)(arow + k0);
        const v8h ahi = *(const v8h*)(arow + k0 + 16);
        v16h av;
#pragma unroll
        for (int e = 0; e < 8; ++e) { av[e] = alo[e]; av[e + 8] = ahi[e]; }
        if (k0 + 256 < K) __builtin_prefetch(arow + k0 + 256, 0, 1);

        // my async transfer for this chunk has landed; barrier makes the
        // whole tile (written by all waves) visible
        asm volatile("s_wait_asynccnt 0x0" ::: "memory");
        __syncthreads();

        v16h bv[NT];
#pragma unroll
        for (int j = 0; j < NT; ++j) {
            const _Float16* bp = &wtile[(j * 16 + mr) * 32 + half * 8];
            const v8h blo = *(const v8h*)(bp);
            const v8h bhi = *(const v8h*)(bp + 16);
#pragma unroll
            for (int e = 0; e < 8; ++e) { bv[j][e] = blo[e]; bv[j][e + 8] = bhi[e]; }
        }
        __syncthreads();   // all waves done reading -> safe to overwrite

        if (producer && (k0 + 32 < K))
            asm volatile("global_load_async_to_lds_b128 %0, %1, off"
                         :: "v"(wdst), "v"(wsrc + k0 + 32) : "memory");

#pragma unroll
        for (int j = 0; j < NT; ++j)
            acc[j] = __builtin_amdgcn_wmma_f32_16x16x32_f16(
                false, av, false, bv[j], (short)0, acc[j], false, false);
    }

    // D layout: VGPR r holds M = r + 8*half, N = n0 + j*16 + (lane&15)
#pragma unroll
    for (int j = 0; j < NT; ++j) {
#pragma unroll
        for (int rr = 0; rr < 8; ++rr) {
            const int m = m0 + rr + 8 * half;
            const int n = n0 + j * 16 + mr;
            float v = acc[j][rr] + ((n < bias_n) ? bias[n] : 0.0f);
            if (RELU_F16)
                ((_Float16*)out)[(size_t)m * N + n] = (_Float16)fmaxf(v, 0.0f);
            else
                ((float*)out)[(size_t)m * N + n] = v;
        }
    }
}

// -------------------------------------------------- log_softmax over 10 cols
__global__ void logsoftmax10_kernel(const float* __restrict__ logits,
                                    float* __restrict__ out) {
    int b = blockIdx.x * blockDim.x + threadIdx.x;
    if (b >= BATCH) return;
    float v[10];
    float mx = -3.4e38f;
#pragma unroll
    for (int i = 0; i < 10; ++i) { v[i] = logits[b * 16 + i]; mx = fmaxf(mx, v[i]); }
    float se = 0.0f;
#pragma unroll
    for (int i = 0; i < 10; ++i) se += __expf(v[i] - mx);
    const float ls = __logf(se) + mx;
#pragma unroll
    for (int i = 0; i < 10; ++i) out[b * 10 + i] = v[i] - ls;
}

// ---------------------------------------------------------------------------
extern "C" void kernel_launch(void* const* d_in, const int* in_sizes, int n_in,
                              void* d_out, int out_size, void* d_ws, size_t ws_size,
                              hipStream_t stream) {
    const float* X   = (const float*)d_in[0];
    const float* w1a = (const float*)d_in[1];
    const float* b1a = (const float*)d_in[2];
    const float* w1b = (const float*)d_in[3];
    const float* b1b = (const float*)d_in[4];
    const float* w2a = (const float*)d_in[5];
    const float* b2a = (const float*)d_in[6];
    const float* w2b = (const float*)d_in[7];
    const float* b2b = (const float*)d_in[8];
    const float* w2c = (const float*)d_in[9];
    const float* b2c = (const float*)d_in[10];
    const float* Wf1 = (const float*)d_in[11];
    const float* bf1 = (const float*)d_in[12];
    const float* Wf2 = (const float*)d_in[13];
    const float* bf2 = (const float*)d_in[14];
    const float* Wf3 = (const float*)d_in[15];
    const float* bf3 = (const float*)d_in[16];
    const float* Wf4 = (const float*)d_in[17];
    const float* bf4 = (const float*)d_in[18];

    // ---- workspace layout (f32 region then f16 region), ~161MB total ----
    float* wsf = (float*)d_ws;
    float* bufA   = wsf;                       // 2048*16*26*26 = 22,151,168
    float* out1   = bufA + 22151168;           // 2048*32*121   =  7,929,856 (reused by conv2b)
    float* out2   = out1 + 7929856;            // 2048*64*16    =  2,097,152
    float* sbuf   = out2 + 2097152;            // 2048*121      =    247,808
    float* logits = sbuf + 247808;             // 2048*16       =     32,768
    _Float16* f0  = (_Float16*)(logits + 32768);  // 16B-aligned
    _Float16* f1  = f0 + (size_t)2048 * 4096;
    _Float16* f2  = f1 + (size_t)2048 * 1024;
    _Float16* f3  = f2 + (size_t)2048 * 256;
    _Float16* W1h = f3 + (size_t)2048 * 64;
    _Float16* W2h = W1h + (size_t)1024 * 4096;
    _Float16* W3h = W2h + (size_t)256 * 1024;
    _Float16* W4h = W3h + (size_t)64 * 256;

    // ---- weight downconversion to f16 ----
    f32_to_f16_kernel<<<(1024 * 4096 + 255) / 256, 256, 0, stream>>>(Wf1, W1h, 1024 * 4096);
    f32_to_f16_kernel<<<(256 * 1024 + 255) / 256, 256, 0, stream>>>(Wf2, W2h, 256 * 1024);
    f32_to_f16_kernel<<<(64 * 256 + 255) / 256, 256, 0, stream>>>(Wf3, W3h, 64 * 256);
    pack_w4_kernel<<<4, 256, 0, stream>>>(Wf4, W4h);

    // ---- conv chains (LDS-staged direct conv, one block per sample) ----
    const size_t sh1 = (size_t)(16 * 26 * 26 + 32 * 16 * 25) * sizeof(float); //  94.5KB
    const size_t sh2 = (size_t)(32 * 11 * 11 + 64 * 32 * 25) * sizeof(float); // 220.3KB
    hipFuncSetAttribute((const void*)conv5x5_s2_kernel,
                        hipFuncAttributeMaxDynamicSharedMemorySize, (int)sh2);

    conv3x3_s1_kernel<<<BATCH, 256, 0, stream>>>(X, w1a, b1a, bufA);
    conv5x5_s2_kernel<<<BATCH, 256, sh1, stream>>>(bufA, w1b, b1b, out1, 16, 26, 32, 11);
    chansum_kernel<<<(BATCH * 121 + 255) / 256, 256, 0, stream>>>(out1, sbuf);

    conv3x3_s1_kernel<<<BATCH, 256, 0, stream>>>(X, w2a, b2a, bufA);
    conv5x5_s2_kernel<<<BATCH, 256, sh1, stream>>>(bufA, w2b, b2b, out1, 16, 26, 32, 11);
    conv5x5_s2_kernel<<<BATCH, 256, sh2, stream>>>(out1, w2c, b2c, out2, 32, 11, 64, 4);

    crosscorr_kernel<<<BATCH, 256, 0, stream>>>(sbuf, out2, f0);

    // ---- FC head on WMMA (M=2048, block tile 128x64) ----
    gemm_wmma_kernel<4, true ><<<dim3(16, 16), 256, 0, stream>>>(f0, W1h, bf1, 1024, f1, 1024, 4096);
    gemm_wmma_kernel<4, true ><<<dim3(16,  4), 256, 0, stream>>>(f1, W2h, bf2,  256, f2,  256, 1024);
    gemm_wmma_kernel<4, true ><<<dim3(16,  1), 256, 0, stream>>>(f2, W3h, bf3,   64, f3,   64,  256);
    gemm_wmma_kernel<1, false><<<dim3(16,  1), 256, 0, stream>>>(f3, W4h, bf4,   10, logits, 16, 64);

    logsoftmax10_kernel<<<(BATCH + 255) / 256, 256, 0, stream>>>(logits, (float*)d_out);
}
